// RNN_33406255628321
// MI455X (gfx1250) — matrix operand order repaired and checked by reading, not compiled, not run on hardware
//
#include <hip/hip_runtime.h>

typedef __attribute__((ext_vector_type(16))) _Float16 v16h;
typedef __attribute__((ext_vector_type(8)))  float    v8f;

#define LOG2E 1.44269504088896340736f

// lane i <-> lane (i ^ 16): VALU-only half-wave swap (no LDS round trip).
static __device__ __forceinline__ float swap16(float v) {
#if __has_builtin(__builtin_amdgcn_permlanex16)
    int vi = __float_as_int(v);
    // identity lane selects: dst lane i reads lane (i%16) of the OTHER half
    int r = __builtin_amdgcn_permlanex16(vi, vi, 0x76543210, 0xfedcba98,
                                         false, false);
    return __int_as_float(r);
#else
    return __shfl_xor(v, 16);
#endif
}

// Persistent single-wave GRU: B=16 batches in the WMMA N dimension, H=8.
// Tile1 (A1): rows 0..15 = W_hh rows 0..15 (r,z gates).
// Tile2 (A2): rows 0..7  = W_hh rows 16..23 (n gate), row 8 = W_out (free y).
// K slots: 0..7 = W_hi x h_hi, 8..15 = W_lo x h_hi, 16..23 = W_hi x h_lo,
//          24..31 = W_lo x h_lo  => full (W_hi+W_lo)*(h_hi+h_lo) product.
__global__ __launch_bounds__(32)
void gru_persistent_wmma(const float* __restrict__ x,
                         const float* __restrict__ W_ih,
                         const float* __restrict__ W_hh,
                         const float* __restrict__ b_ih,
                         const float* __restrict__ b_hh,
                         const float* __restrict__ W_out,
                         const float* __restrict__ b_out,
                         float* __restrict__ y,
                         int T)
{
    const int  lane = threadIdx.x;   // 0..31, one full wave32
    const int  m    = lane & 15;     // A: row index (gate); B/C/D: batch index
    const bool hi   = lane >= 16;    // K-split half / D-row-half

    // ---- Constant A tiles: hi/lo f16 split of the weights (built once) ----
    v16h a1, a2;
#pragma unroll
    for (int k = 0; k < 8; ++k) {
        float w1 = W_hh[m * 8 + k];                    // rows 0..15: r,z
        float w2 = (m < 8)  ? W_hh[(16 + m) * 8 + k]   // rows 0..7 of tile2: n
                 : (m == 8) ? W_out[k] : 0.0f;         // row 8: output proj
        _Float16 w1h = (_Float16)w1;
        _Float16 w2h = (_Float16)w2;
        _Float16 e1 = (_Float16)(w1 - (float)w1h);  // W_lo
        _Float16 e2 = (_Float16)(w2 - (float)w2h);
        a1[k] = w1h; a1[k + 8] = e1;   // elems 0..7: W_hi, elems 8..15: W_lo (both lane halves)
        a2[k] = w2h; a2[k + 8] = e2;
    }

    // ---- Constant C tiles: biases folded into the WMMA accumulator ----
    v8f c1, c2;
#pragma unroll
    for (int j = 0; j < 8; ++j) {
        c1[j] = hi ? (b_ih[8 + j] + b_hh[8 + j])   // D1 rows 8..15: z bias
                   : (b_ih[j]     + b_hh[j]);      // D1 rows 0..7 : r bias
        c2[j] = hi ? (j == 0 ? b_out[0] : 0.0f)    // D2 row 8: y bias
                   : b_hh[16 + j];                 // D2 rows 0..7: n (b_hh only)
    }

    // uniform (SGPR) input-projection weights: IN == 1 so gi = x*w + b
    float wr[8], wz[8], wn[8], bin[8], wo[8];
#pragma unroll
    for (int j = 0; j < 8; ++j) {
        wr[j]  = W_ih[j];
        wz[j]  = W_ih[8 + j];
        wn[j]  = W_ih[16 + j];
        bin[j] = b_ih[16 + j];
        wo[j]  = W_out[j];
    }
    const float bo = b_out[0];

    const float* __restrict__ xb = x + (size_t)m * (size_t)T;
    float*       __restrict__ yb = y + (size_t)m * (size_t)T;

    float h[8];
#pragma unroll
    for (int j = 0; j < 8; ++j) h[j] = 0.0f;

    float xv = xb[0];                      // software-pipelined x stream
    for (int i = 0; i < T; ++i) {
        int nidx = i + 1; if (nidx >= T) nidx = T - 1;
        float xnext = xb[nidx];            // issued early, waited at loop bottom
        if ((i & 31) == 0) {               // uniform branch: stride-ahead prefetch
            int pidx = i + 1024; if (pidx >= T) pidx = T - 1;
            __builtin_prefetch(&xb[pidx], 0, 0);
        }

        // ---- pack B: low lanes carry h_hi (K0..15), high lanes h_lo (K16..31)
        v16h bv;
#pragma unroll
        for (int j = 0; j < 8; ++j) {
            float hoth = swap16(h[j]);                  // high lanes fetch live h
            float hb   = hi ? hoth : h[j];
            _Float16 hh = (_Float16)hb;
            _Float16 he = hi ? (_Float16)(hb - (float)hh) : hh;
            bv[j] = he; bv[j + 8] = he;
        }

        // gh(r,z) and gh(n)+y, biases pre-folded into C
        v8f d1 = __builtin_amdgcn_wmma_f32_16x16x32_f16(
            false, a1, false, bv, (short)0, c1, false, false);
        v8f d2 = __builtin_amdgcn_wmma_f32_16x16x32_f16(
            false, a2, false, bv, (short)0, c2, false, false);

        // D2 row 8 (high lanes, vgpr0) = W_out . h_i + b_out = y[i-1]
        if (i > 0 && hi)
            __builtin_nontemporal_store(d2[0], &yb[i - 1]);

        // ---- gates + state update (valid in low lanes; high lanes masked via B)
#pragma unroll
        for (int j = 0; j < 8; ++j) {
            float zt = swap16(d1[j]);                                 // z rows 8..15
            float ar = fmaf(xv, wr[j], d1[j]);
            float r  = __builtin_amdgcn_rcpf(1.0f + __builtin_amdgcn_exp2f(-LOG2E * ar));
            float az = fmaf(xv, wz[j], zt);
            float z  = __builtin_amdgcn_rcpf(1.0f + __builtin_amdgcn_exp2f(-LOG2E * az));
            float an = fmaf(r, d2[j], fmaf(xv, wn[j], bin[j]));       // tanh arg
            float nn = 1.0f - 2.0f * __builtin_amdgcn_rcpf(
                           1.0f + __builtin_amdgcn_exp2f(2.0f * LOG2E * an));
            h[j] = fmaf(z, h[j] - nn, nn);                            // (1-z)*n + z*h
        }
        xv = xnext;
    }

    // epilogue: y[T-1] = W_out . h_T + b_out
    if (!hi) {
        float acc = bo;
#pragma unroll
        for (int j = 0; j < 8; ++j) acc = fmaf(wo[j], h[j], acc);
        yb[T - 1] = acc;
    }
}

extern "C" void kernel_launch(void* const* d_in, const int* in_sizes, int n_in,
                              void* d_out, int out_size, void* d_ws, size_t ws_size,
                              hipStream_t stream) {
    const float* x     = (const float*)d_in[0];  // (16, 1, T)
    const float* W_ih  = (const float*)d_in[1];  // (24, 1)
    const float* W_hh  = (const float*)d_in[2];  // (24, 8)
    const float* b_ih  = (const float*)d_in[3];  // (24,)
    const float* b_hh  = (const float*)d_in[4];  // (24,)
    const float* W_out = (const float*)d_in[5];  // (1, 8)
    const float* b_out = (const float*)d_in[6];  // (1,)
    float* yp = (float*)d_out;                   // (16, 1, T)

    const int T = in_sizes[0] / 16;              // B*IN*T / B
    gru_persistent_wmma<<<1, 32, 0, stream>>>(x, W_ih, W_hh, b_ih, b_hh,
                                              W_out, b_out, yp, T);
}
